// CRFTagger_4355096838364
// MI455X (gfx1250) — matrix operand order repaired and checked by reading, not compiled, not run on hardware
//
#include <hip/hip_runtime.h>
#include <hip/hip_bf16.h>

typedef __attribute__((ext_vector_type(16))) _Float16 v16h;
typedef __attribute__((ext_vector_type(8)))  float    v8f;

#define CRF_B 1024
#define CRF_T 512
#define CRF_K 64

#define CRF_LOG2E 1.44269504088896340736f
#define CRF_LN2   0.69314718055994530942f

__device__ __forceinline__ v8f wmma_f16(v16h a, v16h b, v8f c) {
  // D(f32 16x16) = A(f16 16x32) * B(f16 32x16) + C
  return __builtin_amdgcn_wmma_f32_16x16x32_f16(
      /*neg_a=*/false, a, /*neg_b=*/false, b,
      /*c_mod=*/(short)0, c, /*reuse_a=*/false, /*reuse_b=*/false);
}

// Load emissions (pre-scaled to log2 units) for one timestep in D-layout order:
// lane (n = lane&15 -> batch col, h = lane>>4), e[m][r] holds tag 16*m + 8*h + r.
__device__ __forceinline__ void load_e(const float* __restrict__ xb, int t, int h,
                                       float e[4][8]) {
#pragma unroll
  for (int m = 0; m < 4; ++m) {
    const float4* p = (const float4*)(xb + (size_t)t * CRF_K + 16 * m + 8 * h);
    float4 q0 = p[0];
    float4 q1 = p[1];
    e[m][0] = q0.x * CRF_LOG2E; e[m][1] = q0.y * CRF_LOG2E;
    e[m][2] = q0.z * CRF_LOG2E; e[m][3] = q0.w * CRF_LOG2E;
    e[m][4] = q1.x * CRF_LOG2E; e[m][5] = q1.y * CRF_LOG2E;
    e[m][6] = q1.z * CRF_LOG2E; e[m][7] = q1.w * CRF_LOG2E;
  }
}

// One forward step (log2 domain): P = exp2(an2) @ exp(trans) via WMMA,
// an2 = log2(P) + e2 - mm, off2 += mm, rebuild B0/B1 = f16(exp2(an2)).
// P in [~0.5, ~100] (each batch column of B holds a 1.0 after normalization),
// so the raw v_log_f32 (no denormal fixup) is safe.
__device__ __forceinline__ void fwd_step(const v16h (&A)[4][2], v16h& B0, v16h& B1,
                                         float& off2, const float e[4][8],
                                         float an[4][8], int h) {
  v8f acc[4];
#pragma unroll
  for (int m = 0; m < 4; ++m) {
    v8f c = {};
    c = wmma_f16(A[m][0], B0, c);
    c = wmma_f16(A[m][1], B1, c);
    acc[m] = c;
  }
  float mm = -__builtin_inff();
#pragma unroll
  for (int m = 0; m < 4; ++m)
#pragma unroll
    for (int r = 0; r < 8; ++r) {
      float v = __builtin_amdgcn_logf(acc[m][r]) + e[m][r];  // log2(P) + e*log2e
      an[m][r] = v;
      mm = fmaxf(mm, v);
    }
  mm = fmaxf(mm, __shfl_xor(mm, 16, 32));  // per-batch max across both halves
  off2 += mm;
#pragma unroll
  for (int m = 0; m < 4; ++m)
#pragma unroll
    for (int r = 0; r < 8; ++r) an[m][r] -= mm;

  // Reassemble next-step B matrices (lane=batch, halfword w -> tag 32*kc+16*h+w).
  // D half h' holds tags {8h'..8h'+7 mod 16}; cross-half values come via shfl_xor(16).
#pragma unroll
  for (int kc = 0; kc < 2; ++kc) {
    v16h nb;
#pragma unroll
    for (int r = 0; r < 8; ++r) {
      float s0 = __shfl_xor(an[2 * kc][r], 16, 32);      // used by h==0 at w=r+8
      float s1 = __shfl_xor(an[2 * kc + 1][r], 16, 32);  // used by h==1 at w=r
      float vlo = h ? s1 : an[2 * kc][r];                // w = r
      float vhi = h ? an[2 * kc + 1][r] : s0;            // w = r + 8
      nb[r]     = (_Float16)__builtin_amdgcn_exp2f(vlo);
      nb[r + 8] = (_Float16)__builtin_amdgcn_exp2f(vhi);
    }
    (kc ? B1 : B0) = nb;
  }
}

__global__ __launch_bounds__(32) void crf_forward_kernel(
    const float* __restrict__ x, const float* __restrict__ trans,
    const float* __restrict__ start, const float* __restrict__ endv,
    float* __restrict__ logZ_out) {
  const int lane = threadIdx.x;
  const int n = lane & 15;   // batch column within tile
  const int h = lane >> 4;   // lane half
  const int b = blockIdx.x * 16 + n;
  const float* xb = x + (size_t)b * CRF_T * CRF_K;

  // Constant A = exp(transitions)^T in WMMA f16 A-layout: A[k_new, k_old].
  v16h A[4][2];
#pragma unroll
  for (int m = 0; m < 4; ++m)
#pragma unroll
    for (int kc = 0; kc < 2; ++kc) {
      v16h a;
#pragma unroll
      for (int w = 0; w < 16; ++w) {
        int kold = 32 * kc + 16 * h + w;
        int knew = 16 * m + n;
        a[w] = (_Float16)__builtin_amdgcn_exp2f(trans[kold * CRF_K + knew] * CRF_LOG2E);
      }
      A[m][kc] = a;
    }

  // Init: alpha0 = start + em[0] (converted to log2 units), normalized B-layout.
  float off2;
  v16h B0, B1;
  {
    float a0[2][16];
#pragma unroll
    for (int kc = 0; kc < 2; ++kc) {
      int tagbase = 32 * kc + 16 * h;
      const float4* p = (const float4*)(xb + tagbase);
#pragma unroll
      for (int q = 0; q < 4; ++q) {
        float4 v = p[q];
        a0[kc][4 * q + 0] = (v.x + start[tagbase + 4 * q + 0]) * CRF_LOG2E;
        a0[kc][4 * q + 1] = (v.y + start[tagbase + 4 * q + 1]) * CRF_LOG2E;
        a0[kc][4 * q + 2] = (v.z + start[tagbase + 4 * q + 2]) * CRF_LOG2E;
        a0[kc][4 * q + 3] = (v.w + start[tagbase + 4 * q + 3]) * CRF_LOG2E;
      }
    }
    float mm = -__builtin_inff();
#pragma unroll
    for (int kc = 0; kc < 2; ++kc)
#pragma unroll
      for (int w = 0; w < 16; ++w) mm = fmaxf(mm, a0[kc][w]);
    mm = fmaxf(mm, __shfl_xor(mm, 16, 32));
    off2 = mm;
#pragma unroll
    for (int kc = 0; kc < 2; ++kc) {
      v16h nb;
#pragma unroll
      for (int w = 0; w < 16; ++w)
        nb[w] = (_Float16)__builtin_amdgcn_exp2f(a0[kc][w] - mm);
      (kc ? B1 : B0) = nb;
    }
  }

  // Forward recursion over t = 1..T-1 with register double-buffered emissions.
  float eA[4][8], eB[4][8];
  float an[4][8];
  load_e(xb, 1, h, eA);
#pragma unroll 1
  for (int t = 1; t < CRF_T; t += 2) {
    int tn = (t + 1 < CRF_T) ? t + 1 : t;
    load_e(xb, tn, h, eB);
    fwd_step(A, B0, B1, off2, eA, an, h);
    if (t + 1 < CRF_T) {
      int tn2 = (t + 2 < CRF_T) ? t + 2 : t + 1;
      load_e(xb, tn2, h, eA);
      fwd_step(A, B0, B1, off2, eB, an, h);
    }
  }

  // logZ = ln2 * (off2 + log2(sum_k exp2(an2_k + end_k*log2e))).
  float s = 0.f;
#pragma unroll
  for (int m = 0; m < 4; ++m)
#pragma unroll
    for (int r = 0; r < 8; ++r) {
      int tag = 16 * m + 8 * h + r;
      s += __builtin_amdgcn_exp2f(an[m][r] + endv[tag] * CRF_LOG2E);
    }
  s += __shfl_xor(s, 16, 32);
  if (h == 0) logZ_out[b] = (off2 + __builtin_amdgcn_logf(s)) * CRF_LN2;
}

__global__ __launch_bounds__(256) void crf_num_kernel(
    const float* __restrict__ x, const int* __restrict__ y,
    const float* __restrict__ trans, const float* __restrict__ start,
    const float* __restrict__ endv, float* __restrict__ num_out) {
  int b = blockIdx.x * blockDim.x + threadIdx.x;
  if (b >= CRF_B) return;
  int prev = y[b];  // y[0, b]
  float acc = start[prev] + x[(size_t)b * CRF_T * CRF_K + prev];
  for (int t = 1; t < CRF_T; ++t) {
    int cur = y[(size_t)t * CRF_B + b];
    acc += trans[prev * CRF_K + cur] + x[((size_t)b * CRF_T + t) * CRF_K + cur];
    prev = cur;
  }
  acc += endv[prev];
  num_out[b] = acc;
}

__global__ __launch_bounds__(256) void crf_reduce_kernel(
    const float* __restrict__ logZ, const float* __restrict__ num,
    float* __restrict__ out) {
  __shared__ float sm[256];
  int tid = threadIdx.x;
  float s = 0.f;
  for (int i = tid; i < CRF_B; i += 256) s += logZ[i] - num[i];
  sm[tid] = s;
  __syncthreads();
  for (int stride = 128; stride > 0; stride >>= 1) {
    if (tid < stride) sm[tid] += sm[tid + stride];
    __syncthreads();
  }
  if (tid == 0) out[0] = sm[0];  // = sum_b (logZ - num) = -(num - logZ).sum()
}

extern "C" void kernel_launch(void* const* d_in, const int* in_sizes, int n_in,
                              void* d_out, int out_size, void* d_ws, size_t ws_size,
                              hipStream_t stream) {
  (void)in_sizes; (void)n_in; (void)out_size; (void)ws_size;
  const float* x     = (const float*)d_in[0];
  // d_in[1] = seq_len (all == T, unused by the reference math)
  const int*   y     = (const int*)d_in[2];
  const float* trans = (const float*)d_in[3];
  const float* start = (const float*)d_in[4];
  const float* endv  = (const float*)d_in[5];

  float* logZ = (float*)d_ws;        // [CRF_B]
  float* num  = logZ + CRF_B;        // [CRF_B]

  hipLaunchKernelGGL(crf_forward_kernel, dim3(CRF_B / 16), dim3(32), 0, stream,
                     x, trans, start, endv, logZ);
  hipLaunchKernelGGL(crf_num_kernel, dim3(CRF_B / 256), dim3(256), 0, stream,
                     x, y, trans, start, endv, num);
  hipLaunchKernelGGL(crf_reduce_kernel, dim3(1), dim3(256), 0, stream,
                     logZ, num, (float*)d_out);
}
